// PatchTST_backbone_TCN_Mix_44461501448324
// MI455X (gfx1250) — matrix operand order repaired and checked by reading, not compiled, not run on hardware
//
#include <hip/hip_runtime.h>
#include <hip/hip_bf16.h>
#include <math.h>

#define TAU 0.75f
#define BROW 200   // im2col LDS row stride in halfs (400B, 16B-aligned, bank-staggered)

typedef __attribute__((ext_vector_type(16))) __bf16 v16bf;
typedef __attribute__((ext_vector_type(8)))  __bf16 v8bf;
typedef __attribute__((ext_vector_type(8)))  float  v8f;

union BF16x16 { v16bf v; v8bf h[2]; unsigned short u[16]; };

__device__ __forceinline__ unsigned short f2bf(float f) {
  unsigned int u = __float_as_uint(f);
  unsigned int r = u + 0x7FFFu + ((u >> 16) & 1u);  // round-to-nearest-even
  return (unsigned short)(r >> 16);
}

// ---------------------------------------------------------------------------
// Kernel 1: fw_chunks (fp32 exact) + bf16 A-fragment pre-swizzle.
// Single block of 256 threads; total work ~5 MFLOP -> negligible.
// ---------------------------------------------------------------------------
__global__ __launch_bounds__(256) void fwchunks_kernel(
    const float* __restrict__ conv_weight,  // [128,64,3]
    const float* __restrict__ bias_param,   // [128]
    const float* __restrict__ grads,        // [24576] -> (64,384)
    const float* __restrict__ W,            // [448,32]
    const float* __restrict__ ctrl_w,       // [64,384]
    const float* __restrict__ ctrl_b,       // [64]
    const float* __restrict__ cw_w,         // [3,64]
    const float* __restrict__ cw_b,         // [3]
    const float* __restrict__ cb_w,         // [2,64]
    const float* __restrict__ cb_b,         // [2]
    const float* __restrict__ cf_w,         // [2,64]
    const float* __restrict__ cf_b,         // [2]
    const int*   __restrict__ trigger,      // [1]
    float* __restrict__ Wupd_out,           // d_out + 32*128*4096, [448*32]
    unsigned short* __restrict__ aw,        // bf16 A frags [8][6][32][16]
    float* __restrict__ bias_eff,           // [128]
    float* __restrict__ f_eff)              // [128]
{
  __shared__ float rep[64 * 64];
  __shared__ float qs[448];     // q = [w(64x3) | b(64x2) | f(64x2)]
  __shared__ float olds[448];   // old_w
  __shared__ float att[32];
  __shared__ float red[256];
  __shared__ int   sidx[2];
  __shared__ float sval[2];
  __shared__ float sden;

  const int t = threadIdx.x;
  const int trig = trigger[0];

  // rep = silu(g @ ctrl_w^T + ctrl_b)  : (64,64)
  for (int e = t; e < 64 * 64; e += 256) {
    int i = e >> 6, j = e & 63;
    const float* gi = grads  + i * 384;
    const float* cj = ctrl_w + j * 384;
    float s = ctrl_b[j];
    for (int k = 0; k < 384; ++k) s += gi[k] * cj[k];
    rep[e] = s / (1.0f + __expf(-s));   // silu
  }
  __syncthreads();

  // q = concat(w.ravel(), b.ravel(), f.ravel())  (448 values)
  for (int e = t; e < 448; e += 256) {
    const float* wt; float bb; int i;
    if (e < 192)      { i = e / 3;           int j = e - 3 * i;   wt = cw_w + j * 64; bb = cw_b[j]; }
    else if (e < 320) { int r = e - 192; i = r >> 1; int j = r & 1; wt = cb_w + j * 64; bb = cb_b[j]; }
    else              { int r = e - 320; i = r >> 1; int j = r & 1; wt = cf_w + j * 64; bb = cf_b[j]; }
    const float* ri = rep + i * 64;
    float s = bb;
    for (int k = 0; k < 64; ++k) s += ri[k] * wt[k];
    qs[e] = s;
  }
  __syncthreads();

  // att = softmax(q @ W / 0.5)  over 32 columns
  if (t < 32) {
    float s = 0.f;
    for (int d = 0; d < 448; ++d) s += qs[d] * W[d * 32 + t];
    att[t] = s * 2.0f;
  }
  __syncthreads();
  if (t < 32) {
    float m = att[0];
    for (int c = 1; c < 32; ++c) m = fmaxf(m, att[c]);
    float sum = 0.f;
    for (int c = 0; c < 32; ++c) sum += __expf(att[c] - m);
    red[t] = __expf(att[t] - m) / sum;
  }
  __syncthreads();
  if (t < 32) att[t] = red[t];
  __syncthreads();

  // top-2 (descending, first index on tie)
  if (t == 0) {
    int i0 = 0;
    for (int c = 1; c < 32; ++c) if (att[c] > att[i0]) i0 = c;
    int i1 = (i0 == 0) ? 1 : 0;
    for (int c = 0; c < 32; ++c) if (c != i0 && att[c] > att[i1]) i1 = c;
    sidx[0] = i0; sidx[1] = i1; sval[0] = att[i0]; sval[1] = att[i1];
  }
  __syncthreads();
  const int   i0 = sidx[0], i1 = sidx[1];
  const float v0 = sval[0], v1 = sval[1];

  // old_w[d] = W[d,i0]*i0 + W[d,i1]*i1
  for (int d = t; d < 448; d += 256)
    olds[d] = W[d * 32 + i0] * (float)i0 + W[d * 32 + i1] * (float)i1;
  __syncthreads();

  // Frobenius norm of updated matrix (pass 1)
  float acc = 0.f;
  for (int e = t; e < 448 * 32; e += 256) {
    int d = e >> 5, j = e & 31;
    float wv = W[e], m;
    if (j == i0)      m = TAU * wv + (1.0f - TAU) * olds[d] * v0;
    else if (j == i1) m = TAU * wv + (1.0f - TAU) * olds[d] * v1;
    else              m = wv;
    acc += m * m;
  }
  red[t] = acc;
  __syncthreads();
  for (int s = 128; s > 0; s >>= 1) {
    if (t < s) red[t] += red[t + s];
    __syncthreads();
  }
  if (t == 0) {
    float n = sqrtf(red[0]);
    sden = 1.0f / (n > 1.0f ? n : 1.0f);   // 1/(relu(n-1)+1)
  }
  __syncthreads();
  const float inv = sden;

  // W_upd (pass 2)
  for (int e = t; e < 448 * 32; e += 256) {
    int d = e >> 5, j = e & 31;
    float wv = W[e], m;
    if (trig == 1) {
      if (j == i0)      m = TAU * wv + (1.0f - TAU) * olds[d] * v0;
      else if (j == i1) m = TAU * wv + (1.0f - TAU) * olds[d] * v1;
      else              m = wv;
      m *= inv;
    } else m = wv;
    Wupd_out[e] = m;
  }

  // bias_eff[co] = bias*b_mixed, f_eff[co] = f_mixed  (co = 2i+j)
  for (int co = t; co < 128; co += 256) {
    int i = co >> 1, j = co & 1;
    float bm = qs[192 + i * 2 + j];
    float fm = qs[320 + i * 2 + j];
    if (trig == 1) {
      bm = TAU * bm + (1.0f - TAU) * olds[192 + i * 2 + j];
      fm = TAU * fm + (1.0f - TAU) * olds[320 + i * 2 + j];
    }
    bias_eff[co] = bias_param[co] * bm;
    f_eff[co]    = fm;
  }

  // bf16 A fragments in WMMA 16-bit A(16x32) register layout:
  // aw[((m*6+s)*32+lane)*16 + j]  holds cw[co=16m+(lane&15)][K=32s+koff]
  // koff = j + (lane>=16?16:0)   (contiguous K order, matches B fragment)
  for (int e = t; e < 8 * 6 * 32 * 16; e += 256) {
    int j    = e & 15;
    int lane = (e >> 4) & 31;
    int s    = (e >> 9) % 6;
    int m    = e / (16 * 32 * 6);
    int co   = 16 * m + (lane & 15);
    int koff = j + ((lane >= 16) ? 16 : 0);
    int kk   = 32 * s + koff;          // kk = ci*3 + k, 0..191
    int ci   = kk / 3, k = kk - 3 * ci;
    float wf = qs[ci * 3 + k];
    if (trig == 1) wf = TAU * wf + (1.0f - TAU) * olds[ci * 3 + k];
    aw[e] = f2bf(conv_weight[(co * 64 + ci) * 3 + k] * wf);
  }
}

// ---------------------------------------------------------------------------
// Kernel 2: implicit-GEMM conv via v_wmma_f32_16x16x32_bf16.
// Block = (batch b, 64-wide L chunk). 8 waves; wave wv owns C_out rows
// 16wv..16wv+15, 4 N-tiles x 6 K-steps (24 WMMAs).
// x staged as bf16 im2col panel xs[nl][kk]: B fragments are contiguous 32B
// runs (2x ds_load_b128); staging reads x via global_load_b128 (4 l at once).
// ---------------------------------------------------------------------------
__global__ __launch_bounds__(256) void conv_wmma_kernel(
    const float* __restrict__ x,            // [32,64,4096]
    const unsigned short* __restrict__ aw,  // bf16 A frags [8][6][32][16]
    const float* __restrict__ bias_eff,     // [128]
    const float* __restrict__ f_eff,        // [128]
    float* __restrict__ out)                // [32,128,4096]
{
  __shared__ unsigned short xs[64 * BROW];  // im2col panel: 64 cols x 192 K (bf16)

  const int blk = blockIdx.x;
  const int b   = blk >> 6;       // 64 L-chunks per batch
  const int l0  = (blk & 63) << 6;
  const int t   = threadIdx.x;

  // Stage im2col: xs[nl*BROW + kk] = bf16(x[b, ci, l0+nl+k-1]), kk = 3*ci+k.
  // Vectorized: each iteration loads 4 consecutive l (global_load_b128,
  // dword-aligned) and stores 4 bf16 columns. Lanes cover consecutive nl4
  // -> one contiguous 256B span per half-wave (16 lanes, one kk).
  const float* xb = x + (size_t)b * 64 * 4096;
  for (int e = t; e < 192 * 16; e += 256) {     // 12 iters/thread
    int kk  = e >> 4;             // 0..191
    int nl4 = (e & 15) << 2;      // 0,4,...,60
    int ci = kk / 3, k = kk - 3 * ci;
    int l = l0 + nl4 + k - 1;
    const float* px = xb + ci * 4096 + l;
    float4 v;
    if (l >= 0 && l + 3 < 4096) {
      v = *(const float4*)px;                   // interior fast path
    } else {                                    // tensor-edge fallback
      v.x = (l + 0 >= 0 && l + 0 < 4096) ? px[0] : 0.0f;
      v.y = (l + 1 >= 0 && l + 1 < 4096) ? px[1] : 0.0f;
      v.z = (l + 2 >= 0 && l + 2 < 4096) ? px[2] : 0.0f;
      v.w = (l + 3 >= 0 && l + 3 < 4096) ? px[3] : 0.0f;
    }
    unsigned short* q = xs + kk;
    q[(nl4 + 0) * BROW] = f2bf(v.x);
    q[(nl4 + 1) * BROW] = f2bf(v.y);
    q[(nl4 + 2) * BROW] = f2bf(v.z);
    q[(nl4 + 3) * BROW] = f2bf(v.w);
  }
  __syncthreads();

  const int wv     = t >> 5;      // wave32 -> wave id = C_out tile
  const int lane   = t & 31;
  const int laneHi = lane >> 4;
  const int n      = lane & 15;   // N column within tile

  // A fragments for this co tile: one contiguous 32B load per K-step
  BF16x16 A[6];
  const unsigned short* ap = aw + ((size_t)(wv * 6) * 32 + lane) * 16;
#pragma unroll
  for (int s = 0; s < 6; ++s)
    A[s].v = *(const v16bf*)(ap + (size_t)s * 32 * 16);

  // per-lane epilogue constants (C/D layout: M = i + 8*laneHi)
  float fe[8], be[8];
#pragma unroll
  for (int i = 0; i < 8; ++i) {
    int co = 16 * wv + i + laneHi * 8;
    fe[i] = f_eff[co];
    be[i] = bias_eff[co];
  }

  for (int nt = 0; nt < 4; ++nt) {
    // B(32x16) layout: laneLo holds K=0..15, laneHi K=16..31 of column n.
    // kk run for step s, lane: 32s + 16*laneHi + (0..15)  -> contiguous in xs.
    const unsigned short* bp = xs + (nt * 16 + n) * BROW + 16 * laneHi;
    v8f c = {};   // f32 accumulator, C=0 (bias applied in epilogue)
#pragma unroll
    for (int s = 0; s < 6; ++s) {
      BF16x16 Bf;
      Bf.h[0] = *(const v8bf*)(bp + 32 * s);      // 16B-aligned ds_load_b128
      Bf.h[1] = *(const v8bf*)(bp + 32 * s + 8);  // 16B-aligned ds_load_b128
      c = __builtin_amdgcn_wmma_f32_16x16x32_bf16(
              false, A[s].v, false, Bf.v, (short)0, c, false, false);
    }
    // out = f * (conv + bias)
    float* ob = out + (size_t)b * 128 * 4096 + (size_t)(l0 + nt * 16 + n);
#pragma unroll
    for (int i = 0; i < 8; ++i) {
      int co = 16 * wv + i + laneHi * 8;
      ob[(size_t)co * 4096] = fe[i] * (c[i] + be[i]);
    }
  }
}

// ---------------------------------------------------------------------------
extern "C" void kernel_launch(void* const* d_in, const int* in_sizes, int n_in,
                              void* d_out, int out_size, void* d_ws, size_t ws_size,
                              hipStream_t stream) {
  const float* x           = (const float*)d_in[0];
  const float* conv_weight = (const float*)d_in[1];
  const float* bias_param  = (const float*)d_in[2];
  const float* grads       = (const float*)d_in[3];
  const float* W           = (const float*)d_in[4];
  const float* ctrl_w      = (const float*)d_in[5];
  const float* ctrl_b      = (const float*)d_in[6];
  const float* cw_w        = (const float*)d_in[7];
  const float* cw_b        = (const float*)d_in[8];
  const float* cb_w        = (const float*)d_in[9];
  const float* cb_b        = (const float*)d_in[10];
  const float* cf_w        = (const float*)d_in[11];
  const float* cf_b        = (const float*)d_in[12];
  const int*   trigger     = (const int*)d_in[13];

  float* out  = (float*)d_out;
  float* Wupd = out + (size_t)32 * 128 * 4096;   // second tuple output

  // workspace: [0,49152) bf16 A fragments; then bias_eff, f_eff (f32)
  unsigned short* aw = (unsigned short*)d_ws;
  float* bias_eff = (float*)((char*)d_ws + 8 * 6 * 32 * 16 * 2);
  float* f_eff    = bias_eff + 128;

  fwchunks_kernel<<<1, 256, 0, stream>>>(
      conv_weight, bias_param, grads, W, ctrl_w, ctrl_b,
      cw_w, cw_b, cb_w, cb_b, cf_w, cf_b, trigger,
      Wupd, aw, bias_eff, f_eff);

  conv_wmma_kernel<<<32 * 64, 256, 0, stream>>>(x, aw, bias_eff, f_eff, out);
}
